// CQFusion_4750233830147
// MI455X (gfx1250) — compile-verified, matrix operand order checked by reading, not run on hardware
//
#include <hip/hip_runtime.h>

#define NEG_INF (-1e30f)

typedef __bf16 bf16;
typedef __attribute__((ext_vector_type(16))) __bf16 v16bf;
typedef __attribute__((ext_vector_type(8)))  float  v8f;

static constexpr int B_  = 16;
static constexpr int LC_ = 2048;
static constexpr int LQ_ = 512;
static constexpr int D_  = 128;
static constexpr int NSPLIT = 8;

__device__ __forceinline__ v8f wmma_bf16(v16bf a, v16bf b, v8f c) {
  // 8 args: (neg_a, A, neg_b, B, c_mod, C, reuse_a, reuse_b)
  return __builtin_amdgcn_wmma_f32_16x16x32_bf16(false, a, false, b, (short)0, c, false, false);
}

__device__ __forceinline__ v8f v8f_zero() {
  v8f z;
#pragma unroll
  for (int i = 0; i < 8; ++i) z[i] = 0.f;
  return z;
}

__device__ __forceinline__ float wave_max(float v) {
#pragma unroll
  for (int off = 16; off >= 1; off >>= 1) v = fmaxf(v, __shfl_xor(v, off, 32));
  return v;
}
__device__ __forceinline__ float wave_sum(float v) {
#pragma unroll
  for (int off = 16; off >= 1; off >>= 1) v += __shfl_xor(v, off, 32);
  return v;
}

// gfx1250 async memory->LDS copy (ASYNCcnt-tracked), per CDNA5 ISA:
//   LDS address taken from the first VGPR operand plus the instruction offset ; global addr from VGPR pair (GV mode)
__device__ __forceinline__ void async_load_b128_to_lds(unsigned lds_off, const void* gaddr) {
  asm volatile("global_load_async_to_lds_b128 %0, %1, off"
               :: "v"(lds_off), "v"(gaddr)
               : "memory");
}
__device__ __forceinline__ void wait_asynccnt0() {
  asm volatile("s_wait_asynccnt 0x0" ::: "memory");
}

// ---------------- prep kernels ----------------

// context -> Ap = bf16(ctx * w4mlu[d]) [B,LC,D]; CbT = bf16(ctx)^T [B,D,LC]
__global__ void k_prep_ctx(const float* __restrict__ ctx, const float* __restrict__ w4mlu,
                           bf16* __restrict__ Ap, bf16* __restrict__ CbT) {
  size_t i = (size_t)blockIdx.x * blockDim.x + threadIdx.x;   // over B*LC*D
  int d = (int)(i % D_);
  size_t bc = i / D_;                                          // b*LC + c
  int b = (int)(bc / LC_);
  int c = (int)(bc % LC_);
  float x = ctx[i];
  Ap[i] = (bf16)(x * w4mlu[d]);
  CbT[((size_t)b * D_ + d) * LC_ + c] = (bf16)x;
}

// query -> Qb = bf16(query) [B,LQ,D]; QbT = bf16(query)^T [B,D,LQ]
__global__ void k_prep_qry(const float* __restrict__ qry,
                           bf16* __restrict__ Qb, bf16* __restrict__ QbT) {
  size_t i = (size_t)blockIdx.x * blockDim.x + threadIdx.x;   // over B*LQ*D
  int d = (int)(i % D_);
  size_t bq = i / D_;
  int b = (int)(bq / LQ_);
  int q = (int)(bq % LQ_);
  float x = qry[i];
  Qb[i] = (bf16)x;
  QbT[((size_t)b * D_ + d) * LQ_ + q] = (bf16)x;
}

// out[row] = X[row,:] . w  (D=128, one wave per row)
__global__ void k_rowdot(const float* __restrict__ X, const float* __restrict__ w,
                         float* __restrict__ out, int nrows) {
  int row  = blockIdx.x * 4 + (threadIdx.x >> 5);
  int lane = threadIdx.x & 31;
  if (row >= nrows) return;
  const float* xr = X + (size_t)row * D_;
  float s = 0.f;
#pragma unroll
  for (int i = 0; i < 4; ++i) { int d = lane + 32 * i; s += xr[d] * w[d]; }
  s = wave_sum(s);
  if (lane == 0) out[row] = s;
}

__global__ void k_cvt_bf16(const float* __restrict__ src, bf16* __restrict__ dst, int n) {
  int i = blockIdx.x * blockDim.x + threadIdx.x;
  if (i < n) dst[i] = (bf16)src[i];
}

// ---------------- score: S = Ap @ Qb^T + cvec + qvec ----------------
// wave computes 16(c) x 64(q); block = 4 waves stacked in c -> 64x64 tile
__global__ void k_score(const bf16* __restrict__ Ap, const bf16* __restrict__ Qb,
                        const float* __restrict__ cvec, const float* __restrict__ qvec,
                        float* __restrict__ S) {
  const int b    = blockIdx.z;
  const int wave = threadIdx.x >> 5;
  const int lane = threadIdx.x & 31;
  const int c0   = blockIdx.x * 64 + wave * 16;
  const int q0   = blockIdx.y * 64;
  const int mrow  = lane & 15;
  const int khalf = (lane >> 4) * 16;

  v8f acc[4];
#pragma unroll
  for (int t = 0; t < 4; ++t) acc[t] = v8f_zero();

  const bf16* arow = Ap + ((size_t)b * LC_ + c0 + mrow) * D_ + khalf;
#pragma unroll
  for (int kk = 0; kk < 4; ++kk) {                      // K = 128, step 32
    v16bf a = *(const v16bf*)(arow + kk * 32);
#pragma unroll
    for (int t = 0; t < 4; ++t) {
      const bf16* brow = Qb + ((size_t)b * LQ_ + q0 + t * 16 + mrow) * D_ + khalf + kk * 32;
      acc[t] = wmma_bf16(a, *(const v16bf*)brow, acc[t]);
    }
  }

  const int n = lane & 15;
  const int mb = (lane >> 4) * 8;
  float cv[8];
#pragma unroll
  for (int j = 0; j < 8; ++j) cv[j] = cvec[(size_t)b * LC_ + c0 + mb + j];
#pragma unroll
  for (int t = 0; t < 4; ++t) {
    float qv = qvec[(size_t)b * LQ_ + q0 + t * 16 + n];
#pragma unroll
    for (int j = 0; j < 8; ++j)
      S[((size_t)b * LC_ + c0 + mb + j) * LQ_ + q0 + t * 16 + n] = acc[t][j] + cv[j] + qv;
  }
}

// ---------------- row softmax over q -> P (bf16) ----------------
__global__ void k_row_softmax(const float* __restrict__ S, const float* __restrict__ q_mask,
                              bf16* __restrict__ P) {
  int row  = blockIdx.x * 4 + (threadIdx.x >> 5);   // b*LC + c
  int lane = threadIdx.x & 31;
  int b = row / LC_;
  const float* srow = S + (size_t)row * LQ_;
  const float* qm   = q_mask + (size_t)b * LQ_;
  float vals[16];
  float m = -3.4e38f;
#pragma unroll
  for (int i = 0; i < 4; ++i) {
    int qb = (lane + 32 * i) * 4;
    float4 v   = *(const float4*)(srow + qb);
    float4 msk = *(const float4*)(qm + qb);
    vals[4*i+0] = v.x + (1.f - msk.x) * NEG_INF;
    vals[4*i+1] = v.y + (1.f - msk.y) * NEG_INF;
    vals[4*i+2] = v.z + (1.f - msk.z) * NEG_INF;
    vals[4*i+3] = v.w + (1.f - msk.w) * NEG_INF;
#pragma unroll
    for (int k2 = 0; k2 < 4; ++k2) m = fmaxf(m, vals[4*i+k2]);
  }
  m = wave_max(m);
  float s = 0.f;
#pragma unroll
  for (int i = 0; i < 16; ++i) { vals[i] = __expf(vals[i] - m); s += vals[i]; }
  s = wave_sum(s);
  float inv = 1.f / s;
#pragma unroll
  for (int i = 0; i < 4; ++i) {
    int qb = (lane + 32 * i) * 4;
#pragma unroll
    for (int k2 = 0; k2 < 4; ++k2)
      P[(size_t)row * LQ_ + qb + k2] = (bf16)(vals[4*i+k2] * inv);
  }
}

// ---------------- column softmax stats (split over c, then combined) ----------------
__global__ void k_col_stats(const float* __restrict__ S, const float* __restrict__ c_mask,
                            float* __restrict__ pm, float* __restrict__ ps) {
  int split = blockIdx.x, b = blockIdx.z;
  int q  = blockIdx.y * 256 + threadIdx.x;
  int c0 = split * (LC_ / NSPLIT);
  float m = -3.4e38f, s = 0.f;
  for (int i = 0; i < LC_ / NSPLIT; ++i) {
    int c = c0 + i;
    float x = S[((size_t)b * LC_ + c) * LQ_ + q] + (1.f - c_mask[(size_t)b * LC_ + c]) * NEG_INF;
    float nm = fmaxf(m, x);
    s = s * __expf(m - nm) + __expf(x - nm);
    m = nm;
  }
  pm[((size_t)b * NSPLIT + split) * LQ_ + q] = m;
  ps[((size_t)b * NSPLIT + split) * LQ_ + q] = s;
}

__global__ void k_col_combine(const float* __restrict__ pm, const float* __restrict__ ps,
                              float* __restrict__ colm, float* __restrict__ cols) {
  int g = blockIdx.x * 256 + threadIdx.x;   // b*LQ + q
  int b = g / LQ_, q = g % LQ_;
  float M = -3.4e38f;
#pragma unroll
  for (int i = 0; i < NSPLIT; ++i) M = fmaxf(M, pm[((size_t)b * NSPLIT + i) * LQ_ + q]);
  float s = 0.f;
#pragma unroll
  for (int i = 0; i < NSPLIT; ++i)
    s += ps[((size_t)b * NSPLIT + i) * LQ_ + q] * __expf(pm[((size_t)b * NSPLIT + i) * LQ_ + q] - M);
  colm[g] = M; cols[g] = s;
}

// ---------------- LDS-transpose + normalize: R[b,q,c] (bf16) ----------------
// Tile staged memory->LDS with gfx1250 async copies (ASYNCcnt), then transposed out.
__global__ void k_transpose_R(const float* __restrict__ S, const float* __restrict__ c_mask,
                              const float* __restrict__ colm, const float* __restrict__ cols,
                              bf16* __restrict__ R) {
  __shared__ float tile[64][68];   // pad 4 floats (16B) so float4 columns dodge bank conflicts
  int b = blockIdx.z, c0 = blockIdx.x * 64, q0 = blockIdx.y * 64;
  int t = threadIdx.x;
  int tr = t >> 4, tc = (t & 15) * 4;
#pragma unroll
  for (int i = 0; i < 4; ++i) {
    int r = tr + i * 16;
    const float* gp = S + ((size_t)b * LC_ + c0 + r) * LQ_ + q0 + tc;
    unsigned loff = (unsigned)(size_t)&tile[r][tc];   // LDS byte offset (addrspace(3))
    async_load_b128_to_lds(loff, gp);                 // 16B memory -> LDS, no VGPR round-trip
  }
  wait_asynccnt0();        // drain this wave's ASYNCcnt before cross-wave barrier
  __syncthreads();
#pragma unroll
  for (int i = 0; i < 4; ++i) {
    int qr = tr + i * 16;
    float M   = colm[(size_t)b * LQ_ + q0 + qr];
    float inv = 1.f / cols[(size_t)b * LQ_ + q0 + qr];
#pragma unroll
    for (int k2 = 0; k2 < 4; ++k2) {
      int c = c0 + tc + k2;
      float x = tile[tc + k2][qr] + (1.f - c_mask[(size_t)b * LC_ + c]) * NEG_INF;
      R[((size_t)b * LQ_ + q0 + qr) * LC_ + c] = (bf16)(__expf(x - M) * inv);
    }
  }
}

// ---------------- T = R @ ctx, stored transposed: TbT [B,D,LQ] (bf16) ----------------
__global__ void k_qT(const bf16* __restrict__ R, const bf16* __restrict__ CbT,
                     bf16* __restrict__ TbT) {
  int b  = blockIdx.y;
  int q0 = blockIdx.x * 16;
  int w = threadIdx.x >> 5, lane = threadIdx.x & 31;
  int d0 = w * 16;
  int mrow = lane & 15, khalf = (lane >> 4) * 16;
  v8f acc = v8f_zero();
  const bf16* arow = R   + ((size_t)b * LQ_ + q0 + mrow) * LC_ + khalf;
  const bf16* brow = CbT + ((size_t)b * D_  + d0 + mrow) * LC_ + khalf;
  for (int c0 = 0; c0 < LC_; c0 += 32) {               // 64 K-steps
    __builtin_prefetch(arow + c0 + 512, 0, 1);
    __builtin_prefetch(brow + c0 + 512, 0, 1);
    v16bf a  = *(const v16bf*)(arow + c0);
    v16bf bb = *(const v16bf*)(brow + c0);
    acc = wmma_bf16(a, bb, acc);
  }
  int n = lane & 15, mb = (lane >> 4) * 8;
#pragma unroll
  for (int j = 0; j < 8; ++j)
    TbT[((size_t)b * D_ + d0 + n) * LQ_ + q0 + mb + j] = (bf16)acc[j];
}

// ---------------- fused c2q = P@query, q2c = P@T ----------------
__global__ void k_pv(const bf16* __restrict__ P, const bf16* __restrict__ QbT,
                     const bf16* __restrict__ TbT,
                     bf16* __restrict__ c2q, bf16* __restrict__ q2c) {
  int b = blockIdx.y, c0 = blockIdx.x * 16;
  int w = threadIdx.x >> 5, lane = threadIdx.x & 31;
  int d0 = w * 16, mrow = lane & 15, khalf = (lane >> 4) * 16;
  v8f acc1 = v8f_zero(), acc2 = v8f_zero();
  const bf16* arow = P   + ((size_t)b * LC_ + c0 + mrow) * LQ_ + khalf;
  const bf16* b1   = QbT + ((size_t)b * D_  + d0 + mrow) * LQ_ + khalf;
  const bf16* b2   = TbT + ((size_t)b * D_  + d0 + mrow) * LQ_ + khalf;
#pragma unroll 4
  for (int q0 = 0; q0 < LQ_; q0 += 32) {               // 16 K-steps
    v16bf a = *(const v16bf*)(arow + q0);
    acc1 = wmma_bf16(a, *(const v16bf*)(b1 + q0), acc1);
    acc2 = wmma_bf16(a, *(const v16bf*)(b2 + q0), acc2);
  }
  int n = lane & 15, mb = (lane >> 4) * 8;
#pragma unroll
  for (int j = 0; j < 8; ++j) {
    size_t o = ((size_t)b * LC_ + c0 + mb + j) * D_ + d0 + n;
    c2q[o] = (bf16)acc1[j];
    q2c[o] = (bf16)acc2[j];
  }
}

// ---------------- final: out = concat(ctx,c2q,ctx*c2q,ctx*q2c) @ W^T + b, * c_mask ----------------
__global__ void k_final(const float* __restrict__ ctx, const bf16* __restrict__ c2q,
                        const bf16* __restrict__ q2c, const bf16* __restrict__ Wb,
                        const float* __restrict__ bias, const float* __restrict__ c_mask,
                        float* __restrict__ out) {
  int b = blockIdx.z;
  int wave = threadIdx.x >> 5, lane = threadIdx.x & 31;
  int c0 = blockIdx.x * 64 + wave * 16;
  int o0 = blockIdx.y * 16;
  int mrow = lane & 15, khalf = (lane >> 4) * 16;
  size_t rowbase = ((size_t)b * LC_ + c0 + mrow) * D_;
  v8f acc = v8f_zero();
#pragma unroll
  for (int kk = 0; kk < 16; ++kk) {                    // K = 512, step 32
    int j0 = kk * 32 + khalf;
    int r = j0 >> 7, jr = j0 & 127;
    v16bf a;
    if (r == 0) {
#pragma unroll
      for (int i = 0; i < 16; ++i) a[i] = (bf16)ctx[rowbase + jr + i];
    } else if (r == 1) {
      a = *(const v16bf*)(c2q + rowbase + jr);
    } else if (r == 2) {
#pragma unroll
      for (int i = 0; i < 16; ++i)
        a[i] = (bf16)(ctx[rowbase + jr + i] * (float)c2q[rowbase + jr + i]);
    } else {
#pragma unroll
      for (int i = 0; i < 16; ++i)
        a[i] = (bf16)(ctx[rowbase + jr + i] * (float)q2c[rowbase + jr + i]);
    }
    v16bf wv = *(const v16bf*)(Wb + (size_t)(o0 + mrow) * (4 * D_) + j0);
    acc = wmma_bf16(a, wv, acc);
  }
  int n = lane & 15, mb = (lane >> 4) * 8;
  float bo = bias[o0 + n];
#pragma unroll
  for (int j = 0; j < 8; ++j) {
    int c = c0 + mb + j;
    out[((size_t)b * LC_ + c) * D_ + o0 + n] = (acc[j] + bo) * c_mask[(size_t)b * LC_ + c];
  }
}

// ---------------- launcher ----------------
extern "C" void kernel_launch(void* const* d_in, const int* in_sizes, int n_in,
                              void* d_out, int out_size, void* d_ws, size_t ws_size,
                              hipStream_t stream) {
  const float* context = (const float*)d_in[0];
  const float* query   = (const float*)d_in[1];
  /* d_in[2] = bridge: unused by the reference */
  const float* c_mask  = (const float*)d_in[3];
  const float* q_mask  = (const float*)d_in[4];
  const float* w4C     = (const float*)d_in[5];
  const float* w4Q     = (const float*)d_in[6];
  const float* w4mlu   = (const float*)d_in[7];
  const float* W       = (const float*)d_in[8];
  const float* bvec    = (const float*)d_in[9];
  float* out = (float*)d_out;

  char* wp = (char*)d_ws;
  auto alloc = [&](size_t bytes) -> void* {
    void* p = (void*)wp;
    wp += (bytes + 255) & ~(size_t)255;
    return p;
  };

  float* S    = (float*)alloc((size_t)B_ * LC_ * LQ_ * 4);   // 67 MB (L2-resident)
  bf16*  P    = (bf16*) alloc((size_t)B_ * LC_ * LQ_ * 2);   // 33.5 MB
  bf16*  R    = (bf16*) alloc((size_t)B_ * LQ_ * LC_ * 2);   // 33.5 MB
  bf16*  Ap   = (bf16*) alloc((size_t)B_ * LC_ * D_ * 2);
  bf16*  CbT  = (bf16*) alloc((size_t)B_ * D_ * LC_ * 2);
  bf16*  Qb   = (bf16*) alloc((size_t)B_ * LQ_ * D_ * 2);
  bf16*  QbT  = (bf16*) alloc((size_t)B_ * D_ * LQ_ * 2);
  bf16*  TbT  = (bf16*) alloc((size_t)B_ * D_ * LQ_ * 2);
  bf16*  c2qB = (bf16*) alloc((size_t)B_ * LC_ * D_ * 2);
  bf16*  q2cB = (bf16*) alloc((size_t)B_ * LC_ * D_ * 2);
  bf16*  Wb   = (bf16*) alloc((size_t)D_ * 4 * D_ * 2);
  float* cvec = (float*)alloc((size_t)B_ * LC_ * 4);
  float* qvec = (float*)alloc((size_t)B_ * LQ_ * 4);
  float* colm = (float*)alloc((size_t)B_ * LQ_ * 4);
  float* cols = (float*)alloc((size_t)B_ * LQ_ * 4);
  float* pm   = (float*)alloc((size_t)B_ * NSPLIT * LQ_ * 4);
  float* ps   = (float*)alloc((size_t)B_ * NSPLIT * LQ_ * 4);

  k_prep_ctx  <<<(B_ * LC_ * D_) / 256, 256, 0, stream>>>(context, w4mlu, Ap, CbT);
  k_prep_qry  <<<(B_ * LQ_ * D_) / 256, 256, 0, stream>>>(query, Qb, QbT);
  k_rowdot    <<<(B_ * LC_) / 4, 128, 0, stream>>>(context, w4C, cvec, B_ * LC_);
  k_rowdot    <<<(B_ * LQ_) / 4, 128, 0, stream>>>(query,   w4Q, qvec, B_ * LQ_);
  k_cvt_bf16  <<<(D_ * 4 * D_) / 256, 256, 0, stream>>>(W, Wb, D_ * 4 * D_);

  k_score     <<<dim3(LC_ / 64, LQ_ / 64, B_), 128, 0, stream>>>(Ap, Qb, cvec, qvec, S);
  k_row_softmax<<<(B_ * LC_) / 4, 128, 0, stream>>>(S, q_mask, P);
  k_col_stats <<<dim3(NSPLIT, LQ_ / 256, B_), 256, 0, stream>>>(S, c_mask, pm, ps);
  k_col_combine<<<(B_ * LQ_) / 256, 256, 0, stream>>>(pm, ps, colm, cols);
  k_transpose_R<<<dim3(LC_ / 64, LQ_ / 64, B_), 256, 0, stream>>>(S, c_mask, colm, cols, R);

  k_qT        <<<dim3(LQ_ / 16, B_), 256, 0, stream>>>(R, CbT, TbT);
  k_pv        <<<dim3(LC_ / 16, B_), 256, 0, stream>>>(P, QbT, TbT, c2qB, q2cB);
  k_final     <<<dim3(LC_ / 64, D_ / 16, B_), 128, 0, stream>>>(context, c2qB, q2cB, Wb, bvec, c_mask, out);

  (void)in_sizes; (void)n_in; (void)out_size; (void)ws_size;
}
